// MessagePassingWithPhase_24043226923414
// MI455X (gfx1250) — matrix operand order repaired and checked by reading, not compiled, not run on hardware
//
#include <hip/hip_runtime.h>
#include <hip/hip_bf16.h>

// MessagePassingWithPhase for MI455X (gfx1250, wave32, WMMA).
// B=2, N=512, D=128, O=4.
// One workgroup per (batch, receiver i). Stream 16-sender tiles with LDS
// double buffering: hidden tile -> LDS (f16), 4x chained
// v_wmma_f32_16x16x32_f16 against W2 (C preloaded with b2), gate via
// v_wmma_f32_16x16x4_f32 (C preloaded with bg), fast-rcp sigmoid epilogue.

#define BB 2
#define NN 512
#define DD 128
#define OO 4

typedef __attribute__((ext_vector_type(16))) _Float16 v16h;
typedef __attribute__((ext_vector_type(8)))  float    v8f;
typedef __attribute__((ext_vector_type(2)))  float    v2f;

// ---------------------------------------------------------------------------
// Kernel 1: base[b,n,:] = X[b,n,:] @ W1r + b1   (recv part, bias folded in)
//           send[b,n,:] = X[b,n,:] @ W1s
// ---------------------------------------------------------------------------
__global__ void __launch_bounds__(128) prelin_kernel(
    const float* __restrict__ X, const float* __restrict__ W1r,
    const float* __restrict__ W1s, const float* __restrict__ b1,
    float* __restrict__ baseO, float* __restrict__ sendO)
{
    __shared__ float xr[DD];
    const int row = blockIdx.x;          // b*N + n
    const int e   = threadIdx.x;
    xr[e] = X[row * DD + e];
    __syncthreads();
    float br = b1[e];
    float bs = 0.0f;
    #pragma unroll 8
    for (int k = 0; k < DD; ++k) {
        const float xv = xr[k];
        br = fmaf(xv, W1r[k * DD + e], br);
        bs = fmaf(xv, W1s[k * DD + e], bs);
    }
    baseO[row * DD + e] = br;
    sendO[row * DD + e] = bs;
}

// ---------------------------------------------------------------------------
// Kernel 2: counts[i] = max(sum_j adj[i,j]!=0, 1)  as float
// ---------------------------------------------------------------------------
__global__ void counts_kernel(const int* __restrict__ adj, float* __restrict__ counts)
{
    const int i = blockIdx.x * blockDim.x + threadIdx.x;
    if (i >= NN) return;
    int c = 0;
    for (int j = 0; j < NN; ++j) c += (adj[i * NN + j] != 0) ? 1 : 0;
    counts[i] = (float)((c > 0) ? c : 1);
}

// ---------------------------------------------------------------------------
// Kernel 3 (main)
// ---------------------------------------------------------------------------
__global__ void __launch_bounds__(256) msg_kernel(
    const float* __restrict__ baseW,   // (B,N,D) recv + b1
    const float* __restrict__ sendW,   // (B,N,D)
    const float* __restrict__ phases,  // (B,N,O)
    const int*   __restrict__ adj,     // (N,N)
    const float* __restrict__ W2,      // (D,D)
    const float* __restrict__ b2,      // (D)
    const float* __restrict__ Wg,      // (O,D)
    const float* __restrict__ bg,      // (D)
    const float* __restrict__ counts,  // (N)
    float* __restrict__ aggW)          // (B,N,D)
{
    __shared__ _Float16 hA[2][16][DD]; // hidden tiles (double buffered), f16
    __shared__ float coh[2][16][OO];   // cos(phi_i - phi_j)
    __shared__ float maskf[2][16];
    __shared__ float baseLDS[DD];      // recv_i + b1
    __shared__ float phiI[OO];

    const int tid    = threadIdx.x;
    const int lane   = tid & 31;
    const int wave   = tid >> 5;       // 0..7 : e-tile index
    const int laneHi = lane >> 4;      // 0 or 1
    const int n16    = lane & 15;

    const int b    = blockIdx.x >> 9;      // / 512
    const int i    = blockIdx.x & (NN - 1);
    const int rowI = b * NN + i;

    if (tid < DD) baseLDS[tid] = baseW[rowI * DD + tid];
    if (tid < OO) phiI[tid]    = phases[rowI * OO + tid];

    // per-lane epilogue constants for output column e = wave*16 + n16
    const int   eg  = wave * 16 + n16;
    const float b2e = b2[eg];
    const float bge = bg[eg];

    // W2 B-fragments: 16-bit B 32x16, this code's contiguous k order. Lanes 0-15 hold K rows 0-15 of
    // the K=32 slab (2 per VGPR), lanes 16-31 hold K rows 16-31.
    v16h bf[4];
    #pragma unroll
    for (int ks = 0; ks < 4; ++ks) {
        const int kb = ks * 32 + laneHi * 16;
        #pragma unroll
        for (int e = 0; e < 16; ++e)
            bf[ks][e] = (_Float16)W2[(kb + e) * DD + eg];
    }

    // Wg B-fragment for the K=4 f32 gate WMMA: lanes 0-15 hold K rows 0-1
    // (1 per VGPR), lanes 16-31 hold K rows 2-3.
    v2f wgB;
    wgB[0] = Wg[(laneHi * 2 + 0) * DD + eg];
    wgB[1] = Wg[(laneHi * 2 + 1) * DD + eg];

    // staging geometry: thread handles 8 contiguous d of one sender row
    const int sjj = tid >> 4;            // sender row in tile
    const int sd  = (tid & 15) * 8;      // d offset, 16B aligned

    auto stage = [&](int jt, int buf) {
        const int j = jt * 16 + sjj;
        const float4 s0 = *(const float4*)&sendW[(b * NN + j) * DD + sd];
        const float4 s1 = *(const float4*)&sendW[(b * NN + j) * DD + sd + 4];
        const float4 c0 = *(const float4*)&baseLDS[sd];
        const float4 c1 = *(const float4*)&baseLDS[sd + 4];
        __align__(16) _Float16 tmp[8];
        tmp[0] = (_Float16)fmaxf(c0.x + s0.x, 0.0f);
        tmp[1] = (_Float16)fmaxf(c0.y + s0.y, 0.0f);
        tmp[2] = (_Float16)fmaxf(c0.z + s0.z, 0.0f);
        tmp[3] = (_Float16)fmaxf(c0.w + s0.w, 0.0f);
        tmp[4] = (_Float16)fmaxf(c1.x + s1.x, 0.0f);
        tmp[5] = (_Float16)fmaxf(c1.y + s1.y, 0.0f);
        tmp[6] = (_Float16)fmaxf(c1.z + s1.z, 0.0f);
        tmp[7] = (_Float16)fmaxf(c1.w + s1.w, 0.0f);
        *(float4*)&hA[buf][sjj][sd] = *(const float4*)tmp;
        if (tid < 64) {
            const int jj = tid >> 2, o = tid & 3;
            coh[buf][jj][o] = __cosf(phiI[o] - phases[(b * NN + jt * 16 + jj) * OO + o]);
        } else if (tid < 80) {
            const int jj = tid - 64;
            maskf[buf][jj] = (adj[i * NN + jt * 16 + jj] != 0) ? 1.0f : 0.0f;
        }
    };

    float acc = 0.0f;

    __syncthreads();          // baseLDS / phiI visible
    stage(0, 0);
    __syncthreads();          // buffer 0 staged

    for (int jt = 0; jt < NN / 16; ++jt) {
        const int buf = jt & 1;
        if (jt + 1 < NN / 16) stage(jt + 1, buf ^ 1);   // overlaps with WMMAs

        // msg tile = hidden(16x128) @ W2(128 x e-tile16), K split in 4x32.
        // C preloaded with b2[e] so the bias add rides the WMMA for free.
        v8f c = {b2e, b2e, b2e, b2e, b2e, b2e, b2e, b2e};
        #pragma unroll
        for (int ks = 0; ks < 4; ++ks) {
            // 16-bit A 16x32, this code's contiguous k order: lane row = n16; lanes>=16 get +16 K offset;
            // 16 contiguous K per lane, 2 consecutive K per VGPR (matches B).
            v16h a;
            const int kb = ks * 32 + laneHi * 16;
            #pragma unroll
            for (int v = 0; v < 8; ++v) {
                const int k = kb + v * 2;
                a[2 * v]     = hA[buf][n16][k];
                a[2 * v + 1] = hA[buf][n16][k + 1];
            }
            c = __builtin_amdgcn_wmma_f32_16x16x32_f16(
                    false, a, false, bf[ks], (short)0, c, false, false);
        }

        // gate pre-activation = coh(16x4) @ Wg(4 x e-tile16) + bg, one K=4
        // f32 WMMA. A layout: lane row = n16, VGPR v = K (lanes>=16: K+2).
        v2f cohA;
        cohA[0] = coh[buf][n16][laneHi * 2 + 0];
        cohA[1] = coh[buf][n16][laneHi * 2 + 1];
        v8f gp = {bge, bge, bge, bge, bge, bge, bge, bge};
        gp = __builtin_amdgcn_wmma_f32_16x16x4_f32(
                 false, cohA, false, wgB, (short)0, gp, false, false);

        // epilogue: element r holds (m = laneHi*8 + r, n = n16)
        #pragma unroll
        for (int r = 0; r < 8; ++r) {
            const int jj = laneHi * 8 + r;
            // fast sigmoid: 1/(1+exp(-g)) via v_rcp_f32
            const float g = __builtin_amdgcn_rcpf(1.0f + __expf(-gp[r]));
            acc = fmaf(maskf[buf][jj] * c[r], g, acc);
        }

        __syncthreads();      // next buffer staged / this buffer reusable
    }

    // merge row-halves: lanes n and n+16 both hold partial column sums of e=eg
    acc += __shfl_xor(acc, 16, 32);
    if (lane < 16) {
        aggW[rowI * DD + eg] = acc / counts[i];
    }
}

// ---------------------------------------------------------------------------
// Kernel 4: update net + residual.
// out[b,n,:] = X + (relu(X@Wu1x + agg@Wu1a + bu1)) @ Wu2 + bu2
// ---------------------------------------------------------------------------
__global__ void __launch_bounds__(128) update_kernel(
    const float* __restrict__ X, const float* __restrict__ agg,
    const float* __restrict__ Wu1x, const float* __restrict__ Wu1a,
    const float* __restrict__ bu1, const float* __restrict__ Wu2,
    const float* __restrict__ bu2, float* __restrict__ out)
{
    __shared__ float xr[DD];
    __shared__ float ar[DD];
    __shared__ float hr[DD];
    const int row = blockIdx.x;
    const int e   = threadIdx.x;
    xr[e] = X[row * DD + e];
    ar[e] = agg[row * DD + e];
    __syncthreads();
    float s = bu1[e];
    #pragma unroll 8
    for (int k = 0; k < DD; ++k) {
        s = fmaf(xr[k], Wu1x[k * DD + e], s);
        s = fmaf(ar[k], Wu1a[k * DD + e], s);
    }
    hr[e] = fmaxf(s, 0.0f);
    __syncthreads();
    float o = bu2[e];
    #pragma unroll 8
    for (int k = 0; k < DD; ++k)
        o = fmaf(hr[k], Wu2[k * DD + e], o);
    out[row * DD + e] = xr[e] + o;
}

// ---------------------------------------------------------------------------
extern "C" void kernel_launch(void* const* d_in, const int* in_sizes, int n_in,
                              void* d_out, int out_size, void* d_ws, size_t ws_size,
                              hipStream_t stream)
{
    const float* X      = (const float*)d_in[0];   // (B,N,D)
    const float* phases = (const float*)d_in[1];   // (B,N,O)
    const int*   adj    = (const int*)  d_in[2];   // (N,N)
    const float* W1r    = (const float*)d_in[3];
    const float* W1s    = (const float*)d_in[4];
    const float* b1     = (const float*)d_in[5];
    const float* W2     = (const float*)d_in[6];
    const float* b2     = (const float*)d_in[7];
    const float* Wg     = (const float*)d_in[8];
    const float* bg     = (const float*)d_in[9];
    const float* Wu1x   = (const float*)d_in[10];
    const float* Wu1a   = (const float*)d_in[11];
    const float* bu1    = (const float*)d_in[12];
    const float* Wu2    = (const float*)d_in[13];
    const float* bu2    = (const float*)d_in[14];
    float* out = (float*)d_out;

    // workspace layout (floats): base | send | agg | counts  (~1.6 MB)
    float* ws     = (float*)d_ws;
    const int RND = BB * NN * DD;       // 131072
    float* baseW  = ws;
    float* sendW  = ws + RND;
    float* aggW   = ws + 2 * RND;
    float* cnts   = ws + 3 * RND;

    prelin_kernel<<<BB * NN, 128, 0, stream>>>(X, W1r, W1s, b1, baseW, sendW);
    counts_kernel<<<(NN + 255) / 256, 256, 0, stream>>>(adj, cnts);
    msg_kernel<<<BB * NN, 256, 0, stream>>>(baseW, sendW, phases, adj,
                                            W2, b2, Wg, bg, cnts, aggW);
    update_kernel<<<BB * NN, 128, 0, stream>>>(X, aggW, Wu1x, Wu1a, bu1,
                                               Wu2, bu2, out);
}
